// AffinePalettizedLinear_51264729645538
// MI455X (gfx1250) — compile-verified, matrix operand order checked by reading, not run on hardware
//
#include <hip/hip_runtime.h>

// ---------------------------------------------------------------------------
// AffinePalettizedLinear on MI455X (gfx1250):
//   Y = X @ LUT[idx]^T + bias,  X:[8192,4096] f32, idx:[16384,4096] i32,
//   LUT:[256] f32, bias:[16384] f32, Y:[8192,16384] f32.
// HBM-bound (~0.9 GB moved vs ~1.1 TFLOP). Fused LUT-dequant bf16 WMMA GEMM:
//  - idx streamed non-temporally (single use), dequantized via LDS LUT
//  - X converted f32->bf16 in the loader; X stays resident in the 192MB L2
//  - grid ordered so blocks sharing a weight tile run together (idx read once)
//  - v_wmma_f32_16x16x32_bf16, 128x128x32 block tile, 8 wave32 waves
// ---------------------------------------------------------------------------

namespace {

constexpr int TOKENS = 8192;
constexpr int IN_F   = 4096;
constexpr int OUT_F  = 16384;

constexpr int BM  = 128;   // token tile
constexpr int BN  = 128;   // out-feature tile
constexpr int BK  = 32;    // K step (matches 16x16x32 bf16 WMMA)
constexpr int BKP = 40;    // padded LDS row pitch (halves): 80B -> no bank conflicts

typedef __attribute__((ext_vector_type(16))) __bf16 v16bf;
typedef __attribute__((ext_vector_type(8)))  __bf16 v8bf;
typedef __attribute__((ext_vector_type(8)))  float  v8f;
typedef __attribute__((ext_vector_type(4)))  int    v4i;   // native vec for NT loads
typedef __attribute__((ext_vector_type(2)))  unsigned int v2u;

// Pack two f32 into two bf16 (round-to-nearest-even), low|high.
__device__ __forceinline__ unsigned int pack2_bf16(float a, float b) {
  unsigned int ua = __float_as_uint(a);
  unsigned int ub = __float_as_uint(b);
  ua += 0x7FFFu + ((ua >> 16) & 1u);
  ub += 0x7FFFu + ((ub >> 16) & 1u);
  return (ua >> 16) | (ub & 0xFFFF0000u);
}

// Build a 16-element bf16 A/B fragment for v_wmma_f32_16x16x32_bf16.
// Per the CDNA5 16-bit A layout: lane holds K=[kg0..kg0+7] in elements 0..7
// and K=[kg0+16..kg0+23] in elements 8..15, with kg0 = (lane<16) ? 0 : 8.
// p points at row start + kg0; both chunks are 16B-aligned (pitch 80B).
__device__ __forceinline__ v16bf ld_frag(const unsigned short* p) {
  v8bf lo = *(const v8bf*)(p);
  v8bf hi = *(const v8bf*)(p + 16);
  v16bf r;
#pragma unroll
  for (int i = 0; i < 8; ++i) { r[i] = lo[i]; r[i + 8] = hi[i]; }
  return r;
}

} // namespace

__global__ __launch_bounds__(256)
void AffinePalettizedLinear_51264729645538_kernel(
    const float* __restrict__ x,     // [TOKENS][IN_F]
    const int*   __restrict__ widx,  // [OUT_F][IN_F]
    const float* __restrict__ lut,   // [256]
    const float* __restrict__ bias,  // [OUT_F]
    float*       __restrict__ out)   // [TOKENS][OUT_F]
{
  __shared__ __align__(16) unsigned short At[BM * BKP]; // X tile (bf16)
  __shared__ __align__(16) unsigned short Bt[BN * BKP]; // dequantized W tile (bf16)
  __shared__ float lutS[256];

  const int tid   = threadIdx.x;        // 256 threads = 8 wave32
  const int lane  = tid & 31;
  const int wave  = tid >> 5;
  const int waveM = wave >> 2;          // 0..1 -> 64 token rows each
  const int waveN = wave & 3;           // 0..3 -> 32 out cols each
  const int m0    = blockIdx.x * BM;    // token dim fastest: blocks sharing a
  const int n0    = blockIdx.y * BN;    // weight tile run concurrently -> idx
                                        // fetched from HBM once, X hits L2
  lutS[tid] = lut[tid];                 // 256 threads == 256 entries
  __syncthreads();

  v8f acc[4][2] = {};                   // 4(m) x 2(n) WMMA tiles per wave

  const int row16 = lane & 15;
  const int kg0   = (lane < 16) ? 0 : 8;

  for (int k0 = 0; k0 < IN_F; k0 += BK) {
    // ---- A tile: X f32 -> bf16 into LDS (4 x float4 per thread) ----
#pragma unroll
    for (int v = 0; v < 4; ++v) {
      int id = tid + v * 256;           // 0..1023 over 128x32 tile
      int r  = id >> 3;                 // row 0..127
      int c4 = (id & 7) << 2;           // k offset 0,4,...,28
      float4 f = *(const float4*)(x + (size_t)(m0 + r) * IN_F + k0 + c4);
      v2u pk;
      pk.x = pack2_bf16(f.x, f.y);
      pk.y = pack2_bf16(f.z, f.w);
      *(v2u*)&At[r * BKP + c4] = pk;
    }
    // ---- B tile: idx (non-temporal) -> LUT dequant -> bf16 into LDS ----
#pragma unroll
    for (int v = 0; v < 4; ++v) {
      int id = tid + v * 256;
      int r  = id >> 3;
      int c4 = (id & 7) << 2;
      v4i q = __builtin_nontemporal_load(
          (const v4i*)(widx + (size_t)(n0 + r) * IN_F + k0 + c4));
      v2u pk;
      pk.x = pack2_bf16(lutS[q[0] & 255], lutS[q[1] & 255]);
      pk.y = pack2_bf16(lutS[q[2] & 255], lutS[q[3] & 255]);
      *(v2u*)&Bt[r * BKP + c4] = pk;
    }
    // Prefetch next K-step lines while this step's WMMAs run.
    if (k0 + BK < IN_F) {
      __builtin_prefetch(widx + (size_t)(n0 + (tid >> 3)) * IN_F + (k0 + BK) + ((tid & 7) << 2), 0, 1);
      __builtin_prefetch(x    + (size_t)(m0 + (tid >> 3)) * IN_F + (k0 + BK) + ((tid & 7) << 2), 0, 1);
    }
    __syncthreads();

    // ---- fragments + 8 WMMAs per wave ----
    v16bf afrag[4], bfrag[2];
#pragma unroll
    for (int i = 0; i < 4; ++i) {
      int r = waveM * 64 + i * 16 + row16;        // token row
      afrag[i] = ld_frag(&At[r * BKP + kg0]);
    }
#pragma unroll
    for (int j = 0; j < 2; ++j) {
      int r = waveN * 32 + j * 16 + row16;        // out-feature row of W
      bfrag[j] = ld_frag(&Bt[r * BKP + kg0]);
    }
#pragma unroll
    for (int i = 0; i < 4; ++i)
#pragma unroll
      for (int j = 0; j < 2; ++j)
        acc[i][j] = __builtin_amdgcn_wmma_f32_16x16x32_bf16(
            /*neg_a=*/false, afrag[i], /*neg_b=*/false, bfrag[j],
            /*c_mod=*/(short)0, acc[i][j], /*reuse_a=*/false, /*reuse_b=*/false);

    __syncthreads();
  }

  // ---- epilogue: bias add + non-temporal store (Y is write-once) ----
  // C/D layout: VGPR r -> M = (lane>=16 ? 8 : 0) + r, N = lane & 15.
  const int colL = lane & 15;
  const int rowH = (lane >> 4) << 3;
#pragma unroll
  for (int j = 0; j < 2; ++j) {
    int n = n0 + waveN * 32 + j * 16 + colL;
    float bv = bias[n];
#pragma unroll
    for (int i = 0; i < 4; ++i) {
      int mb = m0 + waveM * 64 + i * 16 + rowH;
#pragma unroll
      for (int r = 0; r < 8; ++r) {
        __builtin_nontemporal_store(acc[i][j][r] + bv,
                                    out + (size_t)(mb + r) * OUT_F + n);
      }
    }
  }
}

extern "C" void kernel_launch(void* const* d_in, const int* in_sizes, int n_in,
                              void* d_out, int out_size, void* d_ws, size_t ws_size,
                              hipStream_t stream) {
  const float* x    = (const float*)d_in[0];
  const int*   widx = (const int*)d_in[1];
  const float* lut  = (const float*)d_in[2];
  const float* bias = (const float*)d_in[3];
  float* out = (float*)d_out;

  dim3 grid(TOKENS / BM, OUT_F / BN);  // (64, 128); token dim fastest
  AffinePalettizedLinear_51264729645538_kernel<<<grid, dim3(256), 0, stream>>>(
      x, widx, lut, bias, out);
}
